// MTOEModule_618475291034
// MI455X (gfx1250) — compile-verified, matrix operand order checked by reading, not run on hardware
//
#include <hip/hip_runtime.h>
#include <math.h>

#define B_  8
#define S_  4096
#define D_  1024
#define N_  16
#define P_  4
#define H_  2048
#define NP_ 64
#define BS_ (B_*S_)

typedef unsigned short u16;
typedef unsigned int   u32;
typedef __attribute__((ext_vector_type(16))) __bf16 v16bf;
typedef __attribute__((ext_vector_type(8)))  float  v8f;
typedef __attribute__((ext_vector_type(4)))  u32    v4u;
typedef __attribute__((ext_vector_type(8)))  u32    v8u;
typedef int v4i __attribute__((vector_size(16)));   // matches async-LDS builtin param

#define ASG __attribute__((address_space(1)))
#define ASL __attribute__((address_space(3)))

// ---- CDNA5 async global->LDS (guarded) -------------------------------------
#if defined(__has_builtin)
#if __has_builtin(__builtin_amdgcn_global_load_async_to_lds_b128)
#define ASYNC_LDS 1
#endif
#endif

#if defined(ASYNC_LDS)
#if __has_builtin(__builtin_amdgcn_s_wait_asynccnt)
#define WAIT_ASYNC() __builtin_amdgcn_s_wait_asynccnt(0)
#else
#define WAIT_ASYNC() asm volatile("s_wait_asynccnt 0x0" ::: "memory")
#endif
#else
#define WAIT_ASYNC()
#endif

// TDM tensor DMA for the combine-stage LDS staging (inline asm per ISA ch.8)
#define TRY_TDM 1

// ---------- helpers ----------

__device__ __forceinline__ u16 f2bf(float f) {              // f32 -> bf16 RNE
  unsigned u = __float_as_uint(f);
  u += 0x7fffu + ((u >> 16) & 1u);
  return (u16)(u >> 16);
}

union FragU { v16bf v; uint4 q[2]; };

// 16x32 bf16 WMMA fragment (A, or B via pre-transposed B^T) from a row-major
// [16][ld] u16 tile.  ISA 7.12.2: lane&15 = row, lane>>4 = K-half:
// VGPR0-3 = K[h*8..h*8+8), VGPR4-7 = K[16+h*8..).
__device__ __forceinline__ v16bf load_frag(const u16* base, int ld, int lane) {
  int r = lane & 15, h = lane >> 4;
  const u16* p = base + (size_t)r * ld + h * 8;
  FragU f;
  f.q[0] = *(const uint4*)(p);
  f.q[1] = *(const uint4*)(p + 16);
  return f.v;
}

__device__ __forceinline__ v8f wmma_bf16(v16bf a, v16bf b, v8f c) {
  return __builtin_amdgcn_wmma_f32_16x16x32_bf16(false, a, false, b, (short)0, c,
                                                 false, false);
}

// stage nbytes (multiple of 16*blockDim) from global to LDS; caller syncs.
__device__ __forceinline__ void stage_lds(u16* dst, const u16* src, int nbytes) {
  int chunks = nbytes >> 4;
  for (int i = threadIdx.x; i < chunks; i += blockDim.x) {
#if defined(ASYNC_LDS)
    __builtin_amdgcn_global_load_async_to_lds_b128(
        (ASG v4i*)(unsigned long long)(const void*)(src + i * 8),
        (ASL v4i*)(u32)(unsigned long long)(void*)(dst + i * 8), 0, 0);
#else
    *(uint4*)(dst + i * 8) = *(const uint4*)(src + i * 8);
#endif
  }
  WAIT_ASYNC();
}

__device__ __forceinline__ float blk_sum(float v, float* s) {
  int lane = threadIdx.x & 31, w = threadIdx.x >> 5;
#pragma unroll
  for (int o = 16; o > 0; o >>= 1) v += __shfl_xor(v, o, 32);
  if (lane == 0) s[w] = v;
  __syncthreads();
  int nw = (blockDim.x + 31) >> 5;
  if (w == 0) {
    float t = (lane < nw) ? s[lane] : 0.0f;
#pragma unroll
    for (int o = 16; o > 0; o >>= 1) t += __shfl_xor(t, o, 32);
    if (lane == 0) s[0] = t;
  }
  __syncthreads();
  v = s[0];
  __syncthreads();
  return v;
}

__device__ __forceinline__ float blk_max(float v, float* s) {
  int lane = threadIdx.x & 31, w = threadIdx.x >> 5;
#pragma unroll
  for (int o = 16; o > 0; o >>= 1) v = fmaxf(v, __shfl_xor(v, o, 32));
  if (lane == 0) s[w] = v;
  __syncthreads();
  int nw = (blockDim.x + 31) >> 5;
  if (w == 0) {
    float t = (lane < nw) ? s[lane] : -3.4e38f;
#pragma unroll
    for (int o = 16; o > 0; o >>= 1) t = fmaxf(t, __shfl_xor(t, o, 32));
    if (lane == 0) s[0] = t;
  }
  __syncthreads();
  v = s[0];
  __syncthreads();
  return v;
}

// ---------- stage 1: normalization / layout prep ----------

__global__ void k_norm_x(const float* __restrict__ x, u16* __restrict__ xn) {
  __shared__ float sred[8];
  size_t row = blockIdx.x;
  const float* xr = x + row * D_;
  float ss = 0.0f;
  for (int d = threadIdx.x; d < D_; d += blockDim.x) { float v = xr[d]; ss += v * v; }
  ss = blk_sum(ss, sred);
  float inv = 1.0f / fmaxf(sqrtf(ss), 1e-12f);
  u16* orow = xn + row * D_;
  for (int d = threadIdx.x; d < D_; d += blockDim.x) orow[d] = f2bf(xr[d] * inv);
}

__global__ void k_norm_phi(const float* __restrict__ phi, const float* __restrict__ scale,
                           u16* __restrict__ phT) {
  __shared__ float sred[8];
  int np = blockIdx.x;
  float ss = 0.0f;
  for (int d = threadIdx.x; d < D_; d += blockDim.x) {
    float v = phi[(size_t)d * NP_ + np]; ss += v * v;
  }
  ss = blk_sum(ss, sred);
  float inv = scale[0] / fmaxf(sqrtf(ss), 1e-12f);
  for (int d = threadIdx.x; d < D_; d += blockDim.x)
    phT[(size_t)np * D_ + d] = f2bf(phi[(size_t)d * NP_ + np] * inv);
}

__global__ void k_transpose_u16(const u16* __restrict__ src, u16* __restrict__ dst,
                                int R, int C) {
  __shared__ u16 tile[32][33];
  int b = blockIdx.z;
  int r0 = blockIdx.y * 32, c0 = blockIdx.x * 32;
  const u16* s = src + (size_t)b * R * C;
  u16* d = dst + (size_t)b * R * C;
  for (int i = threadIdx.y; i < 32; i += 8)
    tile[i][threadIdx.x] = s[(size_t)(r0 + i) * C + c0 + threadIdx.x];
  __syncthreads();
  for (int i = threadIdx.y; i < 32; i += 8)
    d[(size_t)(c0 + i) * R + r0 + threadIdx.x] = tile[threadIdx.x][i];
}

__global__ void k_conv_transpose(const float* __restrict__ src, u16* __restrict__ dst,
                                 int R, int C) {
  __shared__ u16 tile[32][33];
  int b = blockIdx.z;
  int r0 = blockIdx.y * 32, c0 = blockIdx.x * 32;
  const float* s = src + (size_t)b * R * C;
  u16* d = dst + (size_t)b * R * C;
  for (int i = threadIdx.y; i < 32; i += 8)
    tile[i][threadIdx.x] = f2bf(s[(size_t)(r0 + i) * C + c0 + threadIdx.x]);
  __syncthreads();
  for (int i = threadIdx.y; i < 32; i += 8)
    d[(size_t)(c0 + i) * R + r0 + threadIdx.x] = tile[threadIdx.x][i];
}

// ---------- stage 2: logits GEMM  [BS x 1024] x [1024 x 64] ----------
// B^T (phT, 128 KB) staged in LDS (async); A stream double-buffered:
// two distinct fragment registers (a0/a1), each phase reloads its own buffer.
__global__ void k_logits(const u16* __restrict__ A, const u16* __restrict__ BT,
                         float* __restrict__ C) {
  extern __shared__ __align__(16) char dynsmem[];
  u16* sB = (u16*)dynsmem;                       // 64 x 1024 bf16
  stage_lds(sB, BT, NP_ * D_ * 2);
  __syncthreads();

  int wave = threadIdx.x >> 5, lane = threadIdx.x & 31;
  size_t mtile = (size_t)blockIdx.x * 8 + wave;
  const u16* Ab = A + mtile * 16 * D_;
  v8f acc[4] = {};
  v16bf a0 = load_frag(Ab,      D_, lane);
  v16bf a1 = load_frag(Ab + 32, D_, lane);
  for (int k0 = 0; k0 < D_ - 64; k0 += 64) {
    __builtin_prefetch((const void*)(Ab + k0 + 512), 0, 3);
#pragma unroll
    for (int nt = 0; nt < 4; ++nt)
      acc[nt] = wmma_bf16(a0, load_frag(sB + nt * 16 * D_ + k0, D_, lane), acc[nt]);
    a0 = load_frag(Ab + k0 + 64, D_, lane);
#pragma unroll
    for (int nt = 0; nt < 4; ++nt)
      acc[nt] = wmma_bf16(a1, load_frag(sB + nt * 16 * D_ + k0 + 32, D_, lane), acc[nt]);
    a1 = load_frag(Ab + k0 + 96, D_, lane);
  }
#pragma unroll
  for (int nt = 0; nt < 4; ++nt)
    acc[nt] = wmma_bf16(a0, load_frag(sB + nt * 16 * D_ + (D_ - 64), D_, lane), acc[nt]);
#pragma unroll
  for (int nt = 0; nt < 4; ++nt)
    acc[nt] = wmma_bf16(a1, load_frag(sB + nt * 16 * D_ + (D_ - 32), D_, lane), acc[nt]);

  int n = lane & 15, mb = (lane >> 4) * 8;
#pragma unroll
  for (int nt = 0; nt < 4; ++nt)
#pragma unroll
    for (int v = 0; v < 8; ++v)
      C[(mtile * 16 + mb + v) * NP_ + nt * 16 + n] = acc[nt][v];
}

// ---------- stage 3: softmax statistics ----------

__global__ void k_flat_stats(const float* __restrict__ L, float* __restrict__ fmax,
                             float* __restrict__ fden) {
  __shared__ float sred[32];
  int b = blockIdx.x;
  const float* Lb = L + (size_t)b * S_ * NP_;
  const int n = S_ * NP_;
  float m = -3.4e38f;
  for (int i = threadIdx.x; i < n; i += blockDim.x) m = fmaxf(m, Lb[i]);
  m = blk_max(m, sred);
  float s = 0.0f;
  for (int i = threadIdx.x; i < n; i += blockDim.x) s += expf(Lb[i] - m);
  s = blk_sum(s, sred);
  if (threadIdx.x == 0) { fmax[b] = m; fden[b] = s; }
}

__global__ void k_row_stats(const float* __restrict__ L, const float* __restrict__ fmax,
                            const float* __restrict__ fden, float* __restrict__ p_m,
                            float* __restrict__ p_mt, float* __restrict__ cmb) {
  int wave = threadIdx.x >> 5, lane = threadIdx.x & 31;
  size_t row = (size_t)blockIdx.x * 8 + wave;       // = b*S + s
  int b = (int)(row / S_);
  const float* Lr = L + row * NP_;
  float l0 = Lr[lane], l1 = Lr[lane + 32];
  float rm = fmaxf(l0, l1);
#pragma unroll
  for (int o = 16; o > 0; o >>= 1) rm = fmaxf(rm, __shfl_xor(rm, o, 32));
  float e0 = expf(l0 - rm), e1 = expf(l1 - rm);
  float rs = e0 + e1;
#pragma unroll
  for (int o = 16; o > 0; o >>= 1) rs += __shfl_xor(rs, o, 32);
  cmb[row * NP_ + lane]      = e0 / rs;
  cmb[row * NP_ + lane + 32] = e1 / rs;
  float fm = fmax[b], fd = fden[b];
  float f0 = expf(l0 - fm) / fd, f1 = expf(l1 - fm) / fd;
  float tot = f0 + f1;
#pragma unroll
  for (int o = 16; o > 0; o >>= 1) tot += __shfl_xor(tot, o, 32);
  if (lane == 0) p_m[row] = tot;
  float pp = f0 + f1;                       // p over n (np layout n*4+p)
#pragma unroll
  for (int o = 4; o < 32; o <<= 1) pp += __shfl_xor(pp, o, 32);
  if (lane < 4) p_mt[row * P_ + lane] = pp;
}

__global__ void k_pt(const float* __restrict__ p_mt, float* __restrict__ p_t) {
  __shared__ float sred[8];
  int b = blockIdx.x >> 2, p = blockIdx.x & 3;
  float s = 0.0f;
  for (int i = threadIdx.x; i < S_; i += blockDim.x)
    s += p_mt[((size_t)b * S_ + i) * P_ + p];
  s = blk_sum(s, sred);
  if (threadIdx.x == 0) p_t[blockIdx.x] = s;
}

__global__ void k_mi(const float* __restrict__ p_m, const float* __restrict__ p_mt,
                     const float* __restrict__ p_t, float* __restrict__ partial) {
  __shared__ float sred[8];
  float acc = 0.0f;
  for (size_t row = (size_t)blockIdx.x * blockDim.x + threadIdx.x; row < (size_t)BS_;
       row += (size_t)gridDim.x * blockDim.x) {
    int b = (int)(row / S_);
    float pm = p_m[row];
#pragma unroll
    for (int p = 0; p < P_; ++p) {
      float v = p_mt[row * P_ + p];
      float pt = p_t[b * P_ + p];
      acc += v * logf(v / (pm * pt) + 1e-10f);
    }
  }
  acc = blk_sum(acc, sred);
  if (threadIdx.x == 0) partial[blockIdx.x] = acc;
}

__global__ void k_mi_final(const float* __restrict__ partial, float* __restrict__ out) {
  __shared__ float sred[8];
  float v = (threadIdx.x < 64) ? partial[threadIdx.x] : 0.0f;
  v = blk_sum(v, sred);
  if (threadIdx.x == 0) out[0] = -v;
}

__global__ void k_dsp_stats(const float* __restrict__ L, float* __restrict__ dmax,
                            float* __restrict__ dsum) {
  __shared__ float sred[8];
  int b = blockIdx.x >> 6, np = blockIdx.x & 63;
  const float* Lb = L + (size_t)b * S_ * NP_ + np;
  float m = -3.4e38f;
  for (int i = threadIdx.x; i < S_; i += blockDim.x)
    m = fmaxf(m, Lb[(size_t)i * NP_]);
  m = blk_max(m, sred);
  float s = 0.0f;
  for (int i = threadIdx.x; i < S_; i += blockDim.x)
    s += expf(Lb[(size_t)i * NP_] - m);
  s = blk_sum(s, sred);
  if (threadIdx.x == 0) { dmax[blockIdx.x] = m; dsum[blockIdx.x] = s; }
}

__global__ void k_dspT(const float* __restrict__ L, const float* __restrict__ dmax,
                       const float* __restrict__ dsum, u16* __restrict__ dspT) {
  size_t idx = (size_t)blockIdx.x * blockDim.x + threadIdx.x;   // (b*64+np)*S + s
  int s = (int)(idx & (S_ - 1));
  int bnp = (int)(idx >> 12);
  int np = bnp & 63, b = bnp >> 6;
  float v = expf(L[((size_t)b * S_ + s) * NP_ + np] - dmax[bnp]) / dsum[bnp];
  dspT[idx] = f2bf(v);
}

// ---------- stage 4: xs = dsp^T @ xn   (per b: [64 x 4096] x [4096 x 1024]) ----
// One wave per 16-wide D tile; all 4 M tiles accumulate against each B
// fragment (xnT stream read once).  Explicit 2-phase double buffer, no copies.
__global__ void k_xs_gemm(const u16* __restrict__ dspT, const u16* __restrict__ xnT,
                          const float* __restrict__ temb, float* __restrict__ xs_f,
                          u16* __restrict__ xs_b) {
  int wave = threadIdx.x >> 5, lane = threadIdx.x & 31;
  int b = blockIdx.x >> 3;
  int nt = (blockIdx.x & 7) * 8 + wave;            // 0..63 (D tile)
  const u16* A  = dspT + (size_t)b * NP_ * S_;     // 4 M tiles, stride 16*S
  const u16* Bt = xnT + (size_t)b * D_ * S_ + (size_t)nt * 16 * S_;
  v8f acc[4] = {};
  v16bf aa[2][4], bb[2];
  bb[0] = load_frag(Bt,      S_, lane);
  bb[1] = load_frag(Bt + 32, S_, lane);
#pragma unroll
  for (int mt = 0; mt < 4; ++mt) {
    aa[0][mt] = load_frag(A + (size_t)mt * 16 * S_,      S_, lane);
    aa[1][mt] = load_frag(A + (size_t)mt * 16 * S_ + 32, S_, lane);
  }
  for (int k0 = 0; k0 < S_ - 64; k0 += 64) {
    __builtin_prefetch((const void*)(Bt + k0 + 512), 0, 3);
#pragma unroll
    for (int mt = 0; mt < 4; ++mt) acc[mt] = wmma_bf16(aa[0][mt], bb[0], acc[mt]);
    bb[0] = load_frag(Bt + k0 + 64, S_, lane);
#pragma unroll
    for (int mt = 0; mt < 4; ++mt)
      aa[0][mt] = load_frag(A + (size_t)mt * 16 * S_ + k0 + 64, S_, lane);
#pragma unroll
    for (int mt = 0; mt < 4; ++mt) acc[mt] = wmma_bf16(aa[1][mt], bb[1], acc[mt]);
    bb[1] = load_frag(Bt + k0 + 96, S_, lane);
#pragma unroll
    for (int mt = 0; mt < 4; ++mt)
      aa[1][mt] = load_frag(A + (size_t)mt * 16 * S_ + k0 + 96, S_, lane);
  }
#pragma unroll
  for (int mt = 0; mt < 4; ++mt) acc[mt] = wmma_bf16(aa[0][mt], bb[0], acc[mt]);
#pragma unroll
  for (int mt = 0; mt < 4; ++mt) acc[mt] = wmma_bf16(aa[1][mt], bb[1], acc[mt]);

  int nloc = lane & 15, mb = (lane >> 4) * 8;
  int d = nt * 16 + nloc;
#pragma unroll
  for (int mt = 0; mt < 4; ++mt)
#pragma unroll
    for (int v = 0; v < 8; ++v) {
      int m = mt * 16 + mb + v;                    // np index 0..63
      int ne = m >> 2, p = m & 3;
      float val = acc[mt][v] + temb[(size_t)p * D_ + d];
      size_t o = ((size_t)ne * (B_ * P_) + (b * P_ + p)) * D_ + d;   // [N][BP][D]
      xs_f[o] = val;
      xs_b[o] = f2bf(val);
    }
}

// ---------- stage 5: batched expert MLP ----------
// One wave per 16-wide output tile; both M tiles share each weight fragment
// (weights read once).  Explicit 2-phase double buffer, no copies.
__global__ void k_mlp1(const u16* __restrict__ xs_b, const u16* __restrict__ W1T,
                       const float* __restrict__ b1, u16* __restrict__ h_b) {
  int wave = threadIdx.x >> 5, lane = threadIdx.x & 31;
  int n = blockIdx.x >> 4;
  int nt = (blockIdx.x & 15) * 8 + wave;           // 0..127 (H tile)
  const u16* A  = xs_b + (size_t)n * 32 * D_;
  const u16* Bt = W1T + (size_t)n * H_ * D_ + (size_t)nt * 16 * D_;
  v8f acc[2] = {};
  v16bf aa[2][2], bb[2];
  bb[0] = load_frag(Bt,      D_, lane);
  bb[1] = load_frag(Bt + 32, D_, lane);
#pragma unroll
  for (int mt = 0; mt < 2; ++mt) {
    aa[0][mt] = load_frag(A + (size_t)mt * 16 * D_,      D_, lane);
    aa[1][mt] = load_frag(A + (size_t)mt * 16 * D_ + 32, D_, lane);
  }
  for (int k0 = 0; k0 < D_ - 64; k0 += 64) {
#pragma unroll
    for (int mt = 0; mt < 2; ++mt) acc[mt] = wmma_bf16(aa[0][mt], bb[0], acc[mt]);
    bb[0] = load_frag(Bt + k0 + 64, D_, lane);
#pragma unroll
    for (int mt = 0; mt < 2; ++mt)
      aa[0][mt] = load_frag(A + (size_t)mt * 16 * D_ + k0 + 64, D_, lane);
#pragma unroll
    for (int mt = 0; mt < 2; ++mt) acc[mt] = wmma_bf16(aa[1][mt], bb[1], acc[mt]);
    bb[1] = load_frag(Bt + k0 + 96, D_, lane);
#pragma unroll
    for (int mt = 0; mt < 2; ++mt)
      aa[1][mt] = load_frag(A + (size_t)mt * 16 * D_ + k0 + 96, D_, lane);
  }
#pragma unroll
  for (int mt = 0; mt < 2; ++mt) acc[mt] = wmma_bf16(aa[0][mt], bb[0], acc[mt]);
#pragma unroll
  for (int mt = 0; mt < 2; ++mt) acc[mt] = wmma_bf16(aa[1][mt], bb[1], acc[mt]);

  int nloc = lane & 15, mb = (lane >> 4) * 8;
  int h = nt * 16 + nloc;
#pragma unroll
  for (int mt = 0; mt < 2; ++mt)
#pragma unroll
    for (int v = 0; v < 8; ++v) {
      int m = mt * 16 + mb + v;
      float c = acc[mt][v] + b1[(size_t)n * H_ + h];
      float sv = c / (1.0f + expf(-c));            // silu
      h_b[((size_t)n * 32 + m) * H_ + h] = f2bf(sv);
    }
}

__global__ void k_mlp2(const u16* __restrict__ h_b, const u16* __restrict__ W2T,
                       const float* __restrict__ b2, const float* __restrict__ xs_f,
                       float* __restrict__ r_f) {
  int wave = threadIdx.x >> 5, lane = threadIdx.x & 31;
  int n = blockIdx.x >> 3;
  int nt = (blockIdx.x & 7) * 8 + wave;            // 0..63 (D tile)
  const u16* A  = h_b + (size_t)n * 32 * H_;
  const u16* Bt = W2T + (size_t)n * D_ * H_ + (size_t)nt * 16 * H_;
  v8f acc[2] = {};
  v16bf aa[2][2], bb[2];
  bb[0] = load_frag(Bt,      H_, lane);
  bb[1] = load_frag(Bt + 32, H_, lane);
#pragma unroll
  for (int mt = 0; mt < 2; ++mt) {
    aa[0][mt] = load_frag(A + (size_t)mt * 16 * H_,      H_, lane);
    aa[1][mt] = load_frag(A + (size_t)mt * 16 * H_ + 32, H_, lane);
  }
  for (int k0 = 0; k0 < H_ - 64; k0 += 64) {
#pragma unroll
    for (int mt = 0; mt < 2; ++mt) acc[mt] = wmma_bf16(aa[0][mt], bb[0], acc[mt]);
    bb[0] = load_frag(Bt + k0 + 64, H_, lane);
#pragma unroll
    for (int mt = 0; mt < 2; ++mt)
      aa[0][mt] = load_frag(A + (size_t)mt * 16 * H_ + k0 + 64, H_, lane);
#pragma unroll
    for (int mt = 0; mt < 2; ++mt) acc[mt] = wmma_bf16(aa[1][mt], bb[1], acc[mt]);
    bb[1] = load_frag(Bt + k0 + 96, H_, lane);
#pragma unroll
    for (int mt = 0; mt < 2; ++mt)
      aa[1][mt] = load_frag(A + (size_t)mt * 16 * H_ + k0 + 96, H_, lane);
  }
#pragma unroll
  for (int mt = 0; mt < 2; ++mt) acc[mt] = wmma_bf16(aa[0][mt], bb[0], acc[mt]);
#pragma unroll
  for (int mt = 0; mt < 2; ++mt) acc[mt] = wmma_bf16(aa[1][mt], bb[1], acc[mt]);

  int nloc = lane & 15, mb = (lane >> 4) * 8;
  int d = nt * 16 + nloc;
#pragma unroll
  for (int mt = 0; mt < 2; ++mt)
#pragma unroll
    for (int v = 0; v < 8; ++v) {
      int m = mt * 16 + mb + v;
      size_t o = ((size_t)n * 32 + m) * D_ + d;
      r_f[o] = acc[mt][v] + b2[(size_t)n * D_ + d] + xs_f[o];
    }
}

__global__ void k_ln(const float* __restrict__ r_f, const float* __restrict__ lng,
                     const float* __restrict__ lnb, float* __restrict__ ys) {
  __shared__ float sred[8];
  int row = blockIdx.x;           // n*32 + m   (m = b*4+p)
  int n = row >> 5, m = row & 31;
  int b = m >> 2, p = m & 3;
  const float* rr = r_f + (size_t)row * D_;
  float s = 0.0f, s2 = 0.0f;
  for (int d = threadIdx.x; d < D_; d += blockDim.x) {
    float v = rr[d]; s += v; s2 += v * v;
  }
  s = blk_sum(s, sred);
  s2 = blk_sum(s2, sred);
  float mu = s / D_;
  float var = s2 / D_ - mu * mu;
  float inv = rsqrtf(var + 1e-5f);
  float* orow = ys + (((size_t)(b * P_ + p) * N_) + n) * D_;
  for (int d = threadIdx.x; d < D_; d += blockDim.x)
    orow[d] = (rr[d] - mu) * inv * lng[(size_t)n * D_ + d] + lnb[(size_t)n * D_ + d];
}

// ---------- stage 6: y[p][b][m][d] = sum_n cmb[b][m][n][p] * ys[b][p][n][d] --
// ys[b,p,:,:] (16x1024 f32 = 64 KB) staged in LDS via the Tensor Data Mover.
__global__ void k_combine(const float* __restrict__ ys, const float* __restrict__ cmb,
                          float* __restrict__ out) {
  extern __shared__ __align__(16) char dynsmem[];
  float* yl = (float*)dynsmem;                        // 16 x 1024 f32
  const int chunks = S_ / 64;
  int bp = blockIdx.x / chunks;
  int chunk = blockIdx.x % chunks;
  int b = bp >> 2, p = bp & 3;
  const float* yb = ys + (size_t)bp * N_ * D_;
#if TRY_TDM
  if (threadIdx.x < 32) {                             // one wave issues the DMA
    unsigned long long ga = (unsigned long long)(const void*)yb;
    u32 lds = (u32)(unsigned long long)(void*)yl;
    const u32 elems = N_ * D_;                        // 16384 x 4B, 1-D tile
    v4u g0;
    g0[0] = 1u;                                       // count=1, user mode
    g0[1] = lds;                                      // lds_addr
    g0[2] = (u32)ga;                                  // global_addr[31:0]
    g0[3] = (u32)((ga >> 32) & 0x1FFFFFFull) | (2u << 30);  // addr[56:32], type=2
    v8u g1;
    g1[0] = (2u << 16);                               // wg_mask=0, data_size=4B
    g1[1] = (elems & 0xFFFFu) << 16;                  // tensor_dim0[15:0]
    g1[2] = (elems >> 16) | (1u << 16);               // tensor_dim0[31:16], dim1=1
    g1[3] = (elems & 0xFFFFu) << 16;                  // tile_dim0 = 16384
    g1[4] = 0u;                                       // tile_dim1/2 unused (1-D)
    g1[5] = elems;                                    // tensor_dim0_stride lo
    g1[6] = 0u;
    g1[7] = 0u;
    asm volatile("tensor_load_to_lds %0, %1" :: "s"(g0), "s"(g1) : "memory");
    __builtin_amdgcn_s_wait_tensorcnt(0);
  }
  __syncthreads();
#else
  for (int i = threadIdx.x; i < N_ * D_; i += blockDim.x) yl[i] = yb[i];
  __syncthreads();
#endif
  int d0 = threadIdx.x * 4;
  for (int mi = 0; mi < 64; ++mi) {
    int m = chunk * 64 + mi;
    const float* crow = cmb + ((size_t)b * S_ + m) * NP_ + p;
    float c[N_];
#pragma unroll
    for (int n = 0; n < N_; ++n) c[n] = crow[n * P_];
    float4 a = make_float4(0.f, 0.f, 0.f, 0.f);
#pragma unroll
    for (int n = 0; n < N_; ++n) {
      float4 yv = *(const float4*)&yl[n * D_ + d0];
      a.x += c[n] * yv.x; a.y += c[n] * yv.y;
      a.z += c[n] * yv.z; a.w += c[n] * yv.w;
    }
    *(float4*)&out[(((size_t)p * B_ + b) * S_ + m) * D_ + d0] = a;
  }
}

// ---------- host launcher ----------

extern "C" void kernel_launch(void* const* d_in, const int* in_sizes, int n_in,
                              void* d_out, int out_size, void* d_ws, size_t ws_size,
                              hipStream_t stream) {
  (void)in_sizes; (void)n_in; (void)out_size; (void)ws_size;
  const float* x     = (const float*)d_in[0];
  const float* phi   = (const float*)d_in[1];
  const float* scale = (const float*)d_in[2];
  const float* temb  = (const float*)d_in[3];
  const float* W1    = (const float*)d_in[4];
  const float* b1    = (const float*)d_in[5];
  const float* W2    = (const float*)d_in[6];
  const float* b2    = (const float*)d_in[7];
  const float* lng   = (const float*)d_in[8];
  const float* lnb   = (const float*)d_in[9];

  float* out     = (float*)d_out;
  float* y_out   = out;                                   // 4 * B*S*D
  float* cmb_out = out + (size_t)4 * BS_ * D_;            // B*S*N*P
  float* mi_out  = cmb_out + (size_t)BS_ * NP_;           // 1

  char* w = (char*)d_ws;
  size_t off = 0;
  auto alloc = [&](size_t bytes) -> void* {
    void* p = w + off;
    off = (off + bytes + 255) & ~(size_t)255;
    return p;
  };
  u16*   xn     = (u16*)alloc((size_t)BS_ * D_ * 2);
  u16*   xnT    = (u16*)alloc((size_t)BS_ * D_ * 2);
  u16*   phT    = (u16*)alloc((size_t)NP_ * D_ * 2);
  u16*   W1T    = (u16*)alloc((size_t)N_ * H_ * D_ * 2);
  u16*   W2T    = (u16*)alloc((size_t)N_ * D_ * H_ * 2);
  float* logits = (float*)alloc((size_t)BS_ * NP_ * 4);
  float* fmax   = (float*)alloc(B_ * 4);
  float* fden   = (float*)alloc(B_ * 4);
  float* p_m    = (float*)alloc((size_t)BS_ * 4);
  float* p_mt   = (float*)alloc((size_t)BS_ * P_ * 4);
  float* p_t    = (float*)alloc(B_ * P_ * 4);
  float* mi_par = (float*)alloc(64 * 4);
  float* dmax   = (float*)alloc(B_ * NP_ * 4);
  float* dsum   = (float*)alloc(B_ * NP_ * 4);
  u16*   dspT   = (u16*)alloc((size_t)B_ * NP_ * S_ * 2);
  float* xs_f   = (float*)alloc((size_t)N_ * 32 * D_ * 4);
  u16*   xs_b   = (u16*)alloc((size_t)N_ * 32 * D_ * 2);
  u16*   h_b    = (u16*)alloc((size_t)N_ * 32 * H_ * 2);
  float* r_f    = (float*)alloc((size_t)N_ * 32 * D_ * 4);
  float* ys     = (float*)alloc((size_t)B_ * P_ * N_ * D_ * 4);

  // stage 1
  k_norm_x  <<<BS_, 256, 0, stream>>>(x, xn);
  k_norm_phi<<<NP_, 256, 0, stream>>>(phi, scale, phT);
  {
    dim3 blk(32, 8);
    k_transpose_u16<<<dim3(D_/32, S_/32, B_), blk, 0, stream>>>(xn, xnT, S_, D_);
    k_conv_transpose<<<dim3(H_/32, D_/32, N_), blk, 0, stream>>>(W1, W1T, D_, H_);
    k_conv_transpose<<<dim3(D_/32, H_/32, N_), blk, 0, stream>>>(W2, W2T, H_, D_);
  }
  // stage 2: logits (128 KB dynamic LDS for phT)
  k_logits<<<BS_/16/8, 256, NP_*D_*2, stream>>>(xn, phT, logits);
  // stage 3: softmaxes + MI
  k_flat_stats<<<B_, 1024, 0, stream>>>(logits, fmax, fden);
  k_row_stats <<<BS_/8, 256, 0, stream>>>(logits, fmax, fden, p_m, p_mt, cmb_out);
  k_pt        <<<B_*P_, 256, 0, stream>>>(p_mt, p_t);
  k_mi        <<<64, 256, 0, stream>>>(p_m, p_mt, p_t, mi_par);
  k_mi_final  <<<1, 256, 0, stream>>>(mi_par, mi_out);
  k_dsp_stats <<<B_*NP_, 256, 0, stream>>>(logits, dmax, dsum);
  k_dspT      <<<(B_*NP_*S_)/256, 256, 0, stream>>>(logits, dmax, dsum, dspT);
  // stage 4: xs GEMM
  k_xs_gemm<<<B_*8, 256, 0, stream>>>(dspT, xnT, temb, xs_f, xs_b);
  // stage 5: expert MLP + LN
  k_mlp1<<<N_*16, 256, 0, stream>>>(xs_b, W1T, b1, h_b);
  k_mlp2<<<N_*8, 256, 0, stream>>>(h_b, W2T, b2, xs_f, r_f);
  k_ln  <<<N_*32, 256, 0, stream>>>(r_f, lng, lnb, ys);
  // stage 6: combine -> task features (64 KB dynamic LDS, TDM-staged)
  k_combine<<<B_*P_*(S_/64), 256, N_*D_*4, stream>>>(ys, cmb_out, y_out);
}
